// MultiHeadAttention_35021163331950
// MI455X (gfx1250) — matrix operand order repaired and checked by reading, not compiled, not run on hardware
//
#include <hip/hip_runtime.h>

#define B_ 8
#define T_ 512
#define D_ 1024
#define H_ 16
#define DH_ 64

typedef __attribute__((ext_vector_type(16))) _Float16 v16h;
typedef __attribute__((ext_vector_type(8)))  _Float16 v8h;
typedef __attribute__((ext_vector_type(8)))  float    v8f;

static __device__ __forceinline__ v8f wmma_f32_f16(v16h a, v16h b, v8f c) {
  // D = A(16x32 f16) * B(32x16 f16) + C(16x16 f32)
  return __builtin_amdgcn_wmma_f32_16x16x32_f16(false, a, false, b, (short)0, c,
                                                false, false);
}

// ---- fragment loaders (contiguous k order, shared by A and B) -----------------------
// A (16x32): lane m = tid&15, hf = (tid>>4)&1; elem e -> K = e + 16*hf.
static __device__ __forceinline__ v16h loadA_f32g(const float* __restrict__ src, int ld) {
  const int m  = threadIdx.x & 15;
  const int hf = (threadIdx.x >> 4) & 1;
  const float* p = src + (size_t)m * ld + hf * 16;
  v16h a;
#pragma unroll
  for (int e = 0; e < 8; ++e)  a[e] = (_Float16)p[e];
#pragma unroll
  for (int e = 8; e < 16; ++e) a[e] = (_Float16)p[e];
  return a;
}

static __device__ __forceinline__ v16h loadA_f16(const _Float16* src, int ld) {
  const int m  = threadIdx.x & 15;
  const int hf = (threadIdx.x >> 4) & 1;
  const _Float16* p = src + (size_t)m * ld + hf * 16;
  v16h a;
#pragma unroll
  for (int e = 0; e < 8; ++e)  a[e] = p[e];
#pragma unroll
  for (int e = 8; e < 16; ++e) a[e] = p[e];
  return a;
}

// B (32x16): lane n = tid&15; elem e -> K = e + 16*hf.
// Source "column-major": B[k][n] = src[n*ld + k] (contiguous in k).
static __device__ __forceinline__ v16h loadB_cm(const _Float16* src, int ld) {
  const int n  = threadIdx.x & 15;
  const int hf = (threadIdx.x >> 4) & 1;
  const _Float16* p = src + (size_t)n * ld + hf * 16;
  v16h b;
#pragma unroll
  for (int e = 0; e < 16; ++e) b[e] = p[e];
  return b;
}

static __device__ __forceinline__ v16h loadB_cm_f32(const float* __restrict__ src, int ld) {
  const int n  = threadIdx.x & 15;
  const int hf = (threadIdx.x >> 4) & 1;
  const float* p = src + (size_t)n * ld + hf * 16;
  v16h b;
#pragma unroll
  for (int e = 0; e < 16; ++e) b[e] = (_Float16)p[e];
  return b;
}

// ---- kernel 0: transpose + convert weights to f16 -------------------------
// WqT/WkT/WvT: [H][DH out][DH in] f16 ; WfT: [D out][D in] f16
__global__ __launch_bounds__(256)
void prep_weights(const float* __restrict__ Wq, const float* __restrict__ Wk,
                  const float* __restrict__ Wv, const float* __restrict__ Wf,
                  _Float16* __restrict__ WqT, _Float16* __restrict__ WkT,
                  _Float16* __restrict__ WvT, _Float16* __restrict__ WfT) {
  const int i = blockIdx.x * blockDim.x + threadIdx.x;
  const int NH = H_ * DH_ * DH_;  // 65536
  if (i < NH) {
    int h = i >> 12, r = i & 4095;
    int e = r >> 6, d = r & 63;
    size_t srcI = (size_t)h * 4096 + d * 64 + e;
    WqT[i] = (_Float16)Wq[srcI];
    WkT[i] = (_Float16)Wk[srcI];
    WvT[i] = (_Float16)Wv[srcI];
  }
  const int j = i - NH;
  if (j >= 0 && j < D_ * D_) {
    int e = j >> 10, d = j & 1023;
    WfT[j] = (_Float16)Wf[(size_t)d * D_ + e];
  }
}

// ---- kernel 1: per-(b,h) flash attention with in-LDS K/V ------------------
__global__ __launch_bounds__(256)
void mha_head_kernel(const float* __restrict__ X, const int* __restrict__ mask,
                     const _Float16* __restrict__ WqT, const float* __restrict__ bq,
                     const _Float16* __restrict__ WkT, const float* __restrict__ bk,
                     const _Float16* __restrict__ WvT, const float* __restrict__ bv,
                     _Float16* __restrict__ Hcat) {
  extern __shared__ char smem_raw[];
  _Float16* Ks  = (_Float16*)smem_raw;      // [T_][DH_]  row-major (64 KB)
  _Float16* VsT = Ks + T_ * DH_;            // [DH_][T_]  transposed (64 KB)
  const int wave = threadIdx.x >> 5;
  _Float16* scr = VsT + DH_ * T_ + wave * (16 * DH_);  // per-wave 16x64 f16

  const int b  = blockIdx.x / H_;
  const int h  = blockIdx.x % H_;
  const int n  = threadIdx.x & 15;
  const int hf = (threadIdx.x >> 4) & 1;

  const _Float16* WqTh = WqT + (size_t)h * DH_ * DH_;
  const _Float16* WkTh = WkT + (size_t)h * DH_ * DH_;
  const _Float16* WvTh = WvT + (size_t)h * DH_ * DH_;

  // ---- Phase 1: K^T = WkT * X^T -> Ks[key][feat] ; V = X*Wv -> VsT[feat][key]
  for (int tile = wave; tile < T_ / 16; tile += 8) {
    const int t0 = tile * 16;
    const float* Xb = X + ((size_t)b * T_ + t0) * D_ + h * DH_;
    v16h xb0 = loadB_cm_f32(Xb, D_);      // X^T B-frag, in-feat 0..31
    v16h xb1 = loadB_cm_f32(Xb + 32, D_); // in-feat 32..63
    v16h xa0 = loadA_f32g(Xb, D_);        // X A-frag, in-feat 0..31
    v16h xa1 = loadA_f32g(Xb + 32, D_);
#pragma unroll
    for (int s = 0; s < 4; ++s) {
      v8f ck, cv;
      float bvv = bv[h * DH_ + s * 16 + n];
#pragma unroll
      for (int e = 0; e < 8; ++e) {
        ck[e] = bk[h * DH_ + s * 16 + 8 * hf + e];  // bias per out-feat (row)
        cv[e] = bvv;
      }
      // K^T tile: rows = out-feat, cols = key
      ck = wmma_f32_f16(loadA_f16(WkTh + (s * 16) * DH_, DH_), xb0, ck);
      ck = wmma_f32_f16(loadA_f16(WkTh + (s * 16) * DH_ + 32, DH_), xb1, ck);
      // V tile: rows = key, cols = out-feat
      cv = wmma_f32_f16(xa0, loadB_cm(WvTh + (s * 16) * DH_, DH_), cv);
      cv = wmma_f32_f16(xa1, loadB_cm(WvTh + (s * 16) * DH_ + 32, DH_), cv);
      v8h pk, pv;
#pragma unroll
      for (int e = 0; e < 8; ++e) { pk[e] = (_Float16)ck[e]; pv[e] = (_Float16)cv[e]; }
      *(v8h*)(Ks + (t0 + n) * DH_ + s * 16 + 8 * hf) = pk;   // packed b128
      *(v8h*)(VsT + (s * 16 + n) * T_ + t0 + 8 * hf) = pv;   // packed b128
    }
  }
  __syncthreads();

  // ---- Phase 2: flash attention (all tiles transposed: queries on lanes) --
  const float qscale = 0.125f;  // 1/sqrt(DH_)
  const int* mrow = mask + (size_t)b * T_;
  for (int qt = wave; qt < T_ / 16; qt += 8) {
    const int t0 = qt * 16;
    const float* Xb = X + ((size_t)b * T_ + t0) * D_ + h * DH_;
    v16h xb0 = loadB_cm_f32(Xb, D_);
    v16h xb1 = loadB_cm_f32(Xb + 32, D_);
    // Q^T = WqT * X^T ; staged row-major scr[q][feat] (scale folded in)
#pragma unroll
    for (int s = 0; s < 4; ++s) {
      v8f cq;
#pragma unroll
      for (int e = 0; e < 8; ++e) cq[e] = bq[h * DH_ + s * 16 + 8 * hf + e];
      cq = wmma_f32_f16(loadA_f16(WqTh + (s * 16) * DH_, DH_), xb0, cq);
      cq = wmma_f32_f16(loadA_f16(WqTh + (s * 16) * DH_ + 32, DH_), xb1, cq);
      v8h pq;
#pragma unroll
      for (int e = 0; e < 8; ++e) pq[e] = (_Float16)(cq[e] * qscale);
      *(v8h*)(scr + n * DH_ + s * 16 + 8 * hf) = pq;  // packed b128
    }
    asm volatile("s_wait_dscnt 0" ::: "memory");
    v16h qb0 = loadB_cm(scr, DH_);       // Q^T B-frag (feat 0..31), loop-invariant
    v16h qb1 = loadB_cm(scr + 32, DH_);

    float mi = -3.0e38f, li = 0.0f;      // per-lane (per-query) stats
    v8f acc[4];
#pragma unroll
    for (int s = 0; s < 4; ++s)
#pragma unroll
      for (int e = 0; e < 8; ++e) acc[s][e] = 0.0f;

    for (int jj = 0; jj < T_ / 16; jj += 2) {   // two key tiles per step
      v8f z;
#pragma unroll
      for (int e = 0; e < 8; ++e) z[e] = 0.0f;
      // S^T = K * Q^T : rows = key(e+8hf), cols = query(n)
      const _Float16* K0 = Ks + (jj * 16) * DH_;
      const _Float16* K1 = Ks + ((jj + 1) * 16) * DH_;
      v8f st0 = wmma_f32_f16(loadA_f16(K0, DH_), qb0, z);
      st0     = wmma_f32_f16(loadA_f16(K0 + 32, DH_), qb1, st0);
      v8f st1 = wmma_f32_f16(loadA_f16(K1, DH_), qb0, z);
      st1     = wmma_f32_f16(loadA_f16(K1 + 32, DH_), qb1, st1);
      // mask keys (key index = e + 8*hf within tile); uniform across lanes
#pragma unroll
      for (int e = 0; e < 8; ++e) {
        if (!mrow[jj * 16 + 8 * hf + e])       st0[e] = -1.0e9f;
        if (!mrow[(jj + 1) * 16 + 8 * hf + e]) st1[e] = -1.0e9f;
      }
      // per-lane online softmax over 32 keys (16 local + 16 in other half)
      float vmax = st0[0];
#pragma unroll
      for (int e = 1; e < 8; ++e) vmax = fmaxf(vmax, st0[e]);
#pragma unroll
      for (int e = 0; e < 8; ++e) vmax = fmaxf(vmax, st1[e]);
      vmax = fmaxf(vmax, __shfl_xor(vmax, 16, 32));
      float mn = fmaxf(mi, vmax);
      float alpha = __expf(mi - mn);
      float rs = 0.0f;
      v8h p0, p1;
#pragma unroll
      for (int e = 0; e < 8; ++e) { float t = __expf(st0[e] - mn); rs += t; p0[e] = (_Float16)t; }
#pragma unroll
      for (int e = 0; e < 8; ++e) { float t = __expf(st1[e] - mn); rs += t; p1[e] = (_Float16)t; }
      rs += __shfl_xor(rs, 16, 32);
      li = li * alpha + rs;
      mi = mn;
#pragma unroll
      for (int s = 0; s < 4; ++s)
#pragma unroll
        for (int e = 0; e < 8; ++e) acc[s][e] *= alpha;
      // stage P row-major scr[q][key 0..31] with packed stores
      *(v8h*)(scr + n * 32 + 8 * hf)      = p0;
      *(v8h*)(scr + n * 32 + 16 + 8 * hf) = p1;
      asm volatile("s_wait_dscnt 0" ::: "memory");
      v16h pb = loadB_cm(scr, 32);   // P^T B-frag: B[key][q]
      // O^T += V^T * P^T : rows = feat(e+8hf), cols = query(n)
#pragma unroll
      for (int s = 0; s < 4; ++s)
        acc[s] = wmma_f32_f16(loadA_f16(VsT + (s * 16) * T_ + jj * 16, T_), pb, acc[s]);
    }
    // epilogue: per-lane normalize, packed f16 stores (row = query = lane)
    const float inv = 1.0f / li;
    _Float16* Ob = Hcat + ((size_t)b * T_ + t0 + n) * D_ + h * DH_;
#pragma unroll
    for (int s = 0; s < 4; ++s) {
      v8h o;
#pragma unroll
      for (int e = 0; e < 8; ++e) o[e] = (_Float16)(acc[s][e] * inv);
      *(v8h*)(Ob + s * 16 + 8 * hf) = o;  // packed b128
    }
  }
}

// ---- kernel 2: fusion GEMM (transposed): out^T = WfT * Hcat^T -------------
// WfT slabs staged through LDS with gfx1250 async copy (ASYNCcnt path).
__global__ __launch_bounds__(256)
void fusion_kernel(const _Float16* __restrict__ Hcat, const _Float16* __restrict__ WfT,
                   const float* __restrict__ bf, float* __restrict__ out) {
  __shared__ _Float16 slab[64 * 32];  // [out col 0..63][k 0..31], 4 KB
  const int tid  = threadIdx.x;
  const int wave = tid >> 5;
  const int n    = tid & 15;
  const int hf   = (tid >> 4) & 1;
  const int r0   = blockIdx.x * 128 + wave * 16;  // rows live on lanes now
  const int c0   = blockIdx.y * 64;

  v8f acc[4];
#pragma unroll
  for (int s = 0; s < 4; ++s)
#pragma unroll
    for (int e = 0; e < 8; ++e) acc[s][e] = bf[c0 + s * 16 + 8 * hf + e];

  const int cc = tid >> 2;              // 0..63 : output column within slab
  const int qq = (tid & 3) * 8;         // 0,8,16,24 : k-offset in halfs (16B)
  const unsigned lds_dst = (unsigned)(size_t)(slab + cc * 32 + qq);

  for (int k0 = 0; k0 < D_; k0 += 32) {
    const _Float16* gsrc = WfT + (size_t)(c0 + cc) * D_ + k0 + qq;
    asm volatile("global_load_async_to_lds_b128 %0, %1, off"
                 :: "v"(lds_dst), "v"((unsigned long long)(size_t)gsrc)
                 : "memory");
    asm volatile("s_wait_asynccnt 0" ::: "memory");
    __syncthreads();
    v16h hb = loadB_cm(Hcat + (size_t)r0 * D_ + k0, D_);  // Hcat^T B-frag (invariant over s)
#pragma unroll
    for (int s = 0; s < 4; ++s)
      acc[s] = wmma_f32_f16(loadA_f16(slab + (s * 16) * 32, 32), hb, acc[s]);
    __syncthreads();
  }

  // out rows = lanes -> contiguous f32 stores (8 floats per slice per lane)
  float* Ob = out + (size_t)(r0 + n) * D_ + c0;
#pragma unroll
  for (int s = 0; s < 4; ++s)
    *(v8f*)(Ob + s * 16 + 8 * hf) = acc[s];  // 32B store -> 2x b128
}

extern "C" void kernel_launch(void* const* d_in, const int* in_sizes, int n_in,
                              void* d_out, int out_size, void* d_ws, size_t ws_size,
                              hipStream_t stream) {
  (void)in_sizes; (void)n_in; (void)out_size; (void)ws_size;
  const float* X  = (const float*)d_in[0];
  const int* mask = (const int*)d_in[1];
  const float* Wq = (const float*)d_in[2];
  const float* bq = (const float*)d_in[3];
  const float* Wk = (const float*)d_in[4];
  const float* bk = (const float*)d_in[5];
  const float* Wv = (const float*)d_in[6];
  const float* bv = (const float*)d_in[7];
  const float* Wf = (const float*)d_in[8];
  const float* bf = (const float*)d_in[9];
  float* out = (float*)d_out;

  // ws layout (f16 units): WqT | WkT | WvT | WfT | Hcat
  _Float16* wsh = (_Float16*)d_ws;
  _Float16* WqT = wsh;
  _Float16* WkT = WqT + H_ * DH_ * DH_;
  _Float16* WvT = WkT + H_ * DH_ * DH_;
  _Float16* WfT = WvT + H_ * DH_ * DH_;
  _Float16* Hc  = WfT + (size_t)D_ * D_;   // [B,T,D] f16 (8 MB, L2-resident)

  const int prepN = H_ * DH_ * DH_ + D_ * D_;
  prep_weights<<<dim3((prepN + 255) / 256), dim3(256), 0, stream>>>(
      Wq, Wk, Wv, Wf, WqT, WkT, WvT, WfT);

  const size_t shmem = (size_t)(2 * T_ * DH_ + 8 * 16 * DH_) * sizeof(_Float16); // 147456 B
  mha_head_kernel<<<dim3(B_ * H_), dim3(256), shmem, stream>>>(
      X, mask, WqT, bq, WkT, bk, WvT, bv, Hc);

  fusion_kernel<<<dim3((B_ * T_) / 128, D_ / 64), dim3(256), 0, stream>>>(
      Hc, WfT, bf, out);
}
